// GraphJEPAV0_55190329753875
// MI455X (gfx1250) — compile-verified, matrix operand order checked by reading, not run on hardware
//
#include <hip/hip_runtime.h>
#include <hip/hip_bf16.h>
#include <math.h>

// ---------------- problem constants ----------------
#define K_NODES 32768     // B*TPG
#define K_EDGES 262144    // 8*N
#define K_B     64
#define K_TPG   512
#define K_D     384
#define K_IN    128
#define K_RB    1024
#define K_RD    32
#define K_LD    32
#define K_NH    6
#define K_TP    544       // padded token count (17*32), real T = 513
#define K_BT    (K_B * K_TP)   // 34816 padded token rows
#define EPS_LN  1e-5f

typedef __attribute__((ext_vector_type(16))) _Float16     v16h;
typedef __attribute__((ext_vector_type(8)))  float        v8f;
typedef __attribute__((ext_vector_type(4)))  unsigned int u32x4;
typedef __attribute__((ext_vector_type(8)))  unsigned int u32x8;

// =====================================================================
// Tensor Data Mover: 2D tile copy global -> LDS (f32 elements) with
// hardware padding of 1 dword every 32 dwords (LDS row stride 33).
// D# layout per CDNA5 ISA ch.8 (group0: 4 SGPRs, group1: 8 SGPRs).
// =====================================================================
__device__ __forceinline__ void tdm_load_2d(unsigned int lds_byte, const void* gaddr,
                                            unsigned int tile_d0 /*elems along k*/,
                                            unsigned int tile_d1 /*rows*/,
                                            unsigned int stride_elems,
                                            unsigned int tensor_d0,
                                            unsigned int tensor_d1)
{
    const unsigned long long ga = (unsigned long long)gaddr;
    u32x4 g0;
    g0[0] = 1u;                                              // count=1 (user descriptor)
    g0[1] = lds_byte;                                        // lds_addr
    g0[2] = (unsigned int)ga;                                // global_addr[31:0]
    g0[3] = (unsigned int)((ga >> 32) & 0x01FFFFFFu) | (2u << 30); // addr[56:32] | type=2
    u32x8 g1;
    g1[0] = (2u << 16)        // data_size = 4 bytes
          | (1u << 20)        // pad_enable
          | (4u << 22)        // pad_interval: 32 dwords
          | (0u << 25);       // pad_amount: 1 dword
    g1[1] = (tensor_d0 & 0xFFFFu) << 16;                            // tensor_dim0[15:0]
    g1[2] = ((tensor_d0 >> 16) & 0xFFFFu) | ((tensor_d1 & 0xFFFFu) << 16);
    g1[3] = ((tensor_d1 >> 16) & 0xFFFFu) | ((tile_d0 & 0xFFFFu) << 16); // tile_dim0
    g1[4] = (tile_d1 & 0xFFFFu);                                    // tile_dim1 (tile_dim2=0)
    g1[5] = stride_elems;                                           // tensor_dim0_stride[31:0]
    g1[6] = 0u;                                                     // stride[47:32], dim1_stride lo
    g1[7] = 0u;
    asm volatile("tensor_load_to_lds %0, %1" :: "s"(g0), "s"(g1) : "memory");
}

// =====================================================================
// Generic tiled WMMA GEMM:  C[M,N] = act(A[M,K] @ W[N,K]^T + bias) (+Res)
// Block tile 64x128, 8 waves (2x4), each wave 32x32 via 4 wmma frags.
// A is f32 row-major with row stride lda; W is f32 row-major [N,K].
// Tiles staged by the TDM into LDS (f32, 33-dword padded rows), then
// converted f32->f16 at fragment-build time for v_wmma_f32_16x16x32_f16.
// All M mult of 64, N mult of 128, K mult of 32.
// =====================================================================
#define GBM 64
#define GBN 128
#define GBK 32
#define LDP 33   // padded LDS row stride in dwords (TDM pad: 1 per 32)

template<int ACT, bool RES>
__global__ __launch_bounds__(256)
void k_gemm(const float* __restrict__ A, int lda,
            const float* __restrict__ W,
            const float* __restrict__ bias,
            const float* __restrict__ Res, int ldr,
            float* __restrict__ C, int ldc,
            int M, int N, int K)
{
    __shared__ float As[GBM * LDP];   // [row][k], padded
    __shared__ float Bs[GBN * LDP];   // [n][k],  padded (direct copy of W rows)

    const int tid  = threadIdx.x;
    const int lane = tid & 31;
    const int wv   = tid >> 5;
    const int wrow = (wv >> 2) * 32;   // 0 or 32
    const int wcol = (wv & 3) * 32;    // 0,32,64,96
    const int m0   = blockIdx.y * GBM;
    const int n0   = blockIdx.x * GBN;
    const int hi   = lane >> 4;        // half-wave select
    const int ln   = lane & 15;
    const int kb   = hi * 8;           // K sub-offset per ISA A/B frag layout

    const unsigned int lAs = (unsigned int)(unsigned long long)(&As[0]);
    const unsigned int lBs = (unsigned int)(unsigned long long)(&Bs[0]);

    v8f acc[2][2] = {};

    for (int k0 = 0; k0 < K; k0 += GBK) {
        if (wv == 0) {  // TDM DMA: A tile 64x32, B tile 128x32 (rows of W)
            tdm_load_2d(lAs, A + (size_t)m0 * lda + k0, GBK, GBM,
                        (unsigned)lda, (unsigned)lda, (unsigned)GBM);
            tdm_load_2d(lBs, W + (size_t)n0 * K + k0, GBK, GBN,
                        (unsigned)K, (unsigned)K, (unsigned)GBN);
            __builtin_amdgcn_s_wait_tensorcnt(0);
        }
        __syncthreads();

        if (k0 + GBK < K) {  // gfx1250 global_prefetch_b8 for next tiles
            __builtin_prefetch(&A[(size_t)(m0 + (tid >> 2)) * lda + k0 + GBK], 0, 1);
            __builtin_prefetch(&W[(size_t)(n0 + (tid >> 1)) * K + k0 + GBK], 0, 1);
        }

        #pragma unroll
        for (int mi = 0; mi < 2; ++mi) {
            v16h a;
            const int row = wrow + mi * 16 + ln;   // lanes 0-15 and 16-31 both hold M=ln
            #pragma unroll
            for (int i = 0; i < 8; ++i) {
                a[i]     = (_Float16)As[row * LDP + kb + i];
                a[8 + i] = (_Float16)As[row * LDP + 16 + kb + i];
            }
            #pragma unroll
            for (int ni = 0; ni < 2; ++ni) {
                v16h b;
                const int col = wcol + ni * 16 + ln;
                #pragma unroll
                for (int i = 0; i < 8; ++i) {
                    b[i]     = (_Float16)Bs[col * LDP + kb + i];
                    b[8 + i] = (_Float16)Bs[col * LDP + 16 + kb + i];
                }
                acc[mi][ni] = __builtin_amdgcn_wmma_f32_16x16x32_f16(
                    false, a, false, b, (short)0, acc[mi][ni], false, false);
            }
        }
        __syncthreads();
    }

    // epilogue: C-frag layout -> VGPR j holds row j (+8 for upper half-wave)
    #pragma unroll
    for (int mi = 0; mi < 2; ++mi)
        #pragma unroll
        for (int ni = 0; ni < 2; ++ni)
            #pragma unroll
            for (int j = 0; j < 8; ++j) {
                const int gr = m0 + wrow + mi * 16 + j + hi * 8;
                const int gc = n0 + wcol + ni * 16 + ln;
                float v = acc[mi][ni][j];
                if (bias) v += bias[gc];
                if (ACT == 1) v = fmaxf(v, 0.f);
                if (ACT == 2) v = 0.5f * v * (1.f + erff(v * 0.70710678118f));
                if (RES) v += Res[(size_t)gr * ldr + gc];
                C[(size_t)gr * ldc + gc] = v;
            }
}

// =====================================================================
// Flash-attention: one wave per (graph, head, 16-query tile).
// qkv rows have stride 1152: [Q(384) | K(384) | V(384)], head slice 64.
// Keys processed 32 at a time; cols >= 513 masked; 1/8 folded into Q.
// =====================================================================
__global__ __launch_bounds__(256)
void k_attention(const float* __restrict__ qkv, float* __restrict__ obuf)
{
    __shared__ _Float16 Pst[8][16 * 32];   // per-wave P staging (C-frag -> A-frag relayout)

    const int lane = threadIdx.x & 31;
    const int wv   = threadIdx.x >> 5;
    const int gw   = blockIdx.x * 8 + wv;       // 13056 waves total, exact grid
    const int qt   = gw % 34;                   // query tile (34 tiles of 16 = 544)
    const int bh   = gw / 34;
    const int hh   = bh % K_NH;
    const int b    = bh / K_NH;
    const int hi   = lane >> 4;
    const int ln   = lane & 15;
    const int kb   = hi * 8;

    const float* Qp = qkv + (size_t)b * K_TP * 1152 + hh * 64;
    const float* Kp = Qp + 384;
    const float* Vp = Qp + 768;

    // Q fragments for rows qt*16+ln, d-chunks [0,32) and [32,64); scale folded
    v16h aQ[2];
    {
        const float* qrow = Qp + (size_t)(qt * 16 + ln) * 1152;
        #pragma unroll
        for (int c = 0; c < 2; ++c)
            #pragma unroll
            for (int i = 0; i < 8; ++i) {
                aQ[c][i]     = (_Float16)(qrow[c * 32 + kb + i] * 0.125f);
                aQ[c][8 + i] = (_Float16)(qrow[c * 32 + 16 + kb + i] * 0.125f);
            }
    }

    v8f o[4] = {};                 // O accumulator, d tiles of 16
    float m[8], l[8];
    #pragma unroll
    for (int j = 0; j < 8; ++j) { m[j] = -1e30f; l[j] = 0.f; }

    for (int kt = 0; kt < K_TP; kt += 32) {
        // ---- S = Q @ K^T (16 x 32 as two 16x16 col tiles) ----
        v8f s[2] = {{}, {}};
        #pragma unroll
        for (int nt = 0; nt < 2; ++nt) {
            const float* krow = Kp + (size_t)(kt + nt * 16 + ln) * 1152;
            #pragma unroll
            for (int c = 0; c < 2; ++c) {
                v16h bk;
                #pragma unroll
                for (int i = 0; i < 8; ++i) {
                    bk[i]     = (_Float16)krow[c * 32 + kb + i];
                    bk[8 + i] = (_Float16)krow[c * 32 + 16 + kb + i];
                }
                s[nt] = __builtin_amdgcn_wmma_f32_16x16x32_f16(
                    false, aQ[c], false, bk, (short)0, s[nt], false, false);
            }
        }
        // mask padded key columns (>= 513)
        #pragma unroll
        for (int nt = 0; nt < 2; ++nt) {
            const int col = kt + nt * 16 + ln;
            if (col >= 513)
                #pragma unroll
                for (int j = 0; j < 8; ++j) s[nt][j] = -1e30f;
        }
        // ---- online softmax (row spread over 16 lanes within each half) ----
        float alpha[8];
        #pragma unroll
        for (int j = 0; j < 8; ++j) {
            float mx = fmaxf(s[0][j], s[1][j]);
            #pragma unroll
            for (int off = 1; off < 16; off <<= 1)
                mx = fmaxf(mx, __shfl_xor(mx, off));
            const float mn = fmaxf(m[j], mx);
            const float al = __expf(m[j] - mn);
            m[j] = mn; l[j] *= al; alpha[j] = al;
            const float p0 = __expf(s[0][j] - mn);
            const float p1 = __expf(s[1][j] - mn);
            s[0][j] = p0; s[1][j] = p1;
            float rs = p0 + p1;
            #pragma unroll
            for (int off = 1; off < 16; off <<= 1)
                rs += __shfl_xor(rs, off);
            l[j] += rs;
        }
        #pragma unroll
        for (int dt = 0; dt < 4; ++dt)
            #pragma unroll
            for (int j = 0; j < 8; ++j) o[dt][j] *= alpha[j];

        // ---- relayout P: C-frag -> row-major LDS -> A-frag f16 ----
        _Float16* P = Pst[wv];
        #pragma unroll
        for (int j = 0; j < 8; ++j) {
            P[(j + hi * 8) * 32 + ln]      = (_Float16)s[0][j];
            P[(j + hi * 8) * 32 + 16 + ln] = (_Float16)s[1][j];
        }
        v16h ap;
        #pragma unroll
        for (int i = 0; i < 8; ++i) {
            ap[i]     = P[ln * 32 + kb + i];
            ap[8 + i] = P[ln * 32 + 16 + kb + i];
        }
        // ---- O += P @ V (V tile 32x64 as 4 B-frags) ----
        #pragma unroll
        for (int dt = 0; dt < 4; ++dt) {
            v16h bv;
            #pragma unroll
            for (int i = 0; i < 8; ++i) {
                bv[i]     = (_Float16)Vp[(size_t)(kt + kb + i) * 1152 + dt * 16 + ln];
                bv[8 + i] = (_Float16)Vp[(size_t)(kt + 16 + kb + i) * 1152 + dt * 16 + ln];
            }
            o[dt] = __builtin_amdgcn_wmma_f32_16x16x32_f16(
                false, ap, false, bv, (short)0, o[dt], false, false);
        }
    }

    float* orow = obuf + ((size_t)b * K_TP + qt * 16) * K_D + hh * 64;
    #pragma unroll
    for (int j = 0; j < 8; ++j) {
        const float inv = 1.0f / l[j];
        const int r = j + hi * 8;
        #pragma unroll
        for (int dt = 0; dt < 4; ++dt)
            orow[(size_t)r * K_D + dt * 16 + ln] = o[dt][j] * inv;
    }
}

// =====================================================================
// Small kernels
// =====================================================================
__global__ void k_embed_concat(const float* __restrict__ x, const int* __restrict__ label,
                               const float* __restrict__ lemb, float* __restrict__ out)
{
    const size_t i = (size_t)blockIdx.x * 256 + threadIdx.x;
    if (i >= (size_t)K_NODES * 160) return;
    const int c = (int)(i % 160);
    const size_t node = i / 160;
    out[i] = (c < K_IN) ? x[node * K_IN + c]
                        : lemb[(size_t)label[node] * K_LD + (c - K_IN)];
}

__global__ void k_edge_scatter(const int* __restrict__ src, const int* __restrict__ dst,
                               const int* __restrict__ rel,
                               const float* __restrict__ hn, const float* __restrict__ relW,
                               float* __restrict__ agg, float* __restrict__ cnt)
{
    const int e    = (blockIdx.x * blockDim.x + threadIdx.x) >> 5;   // wave per edge
    const int lane = threadIdx.x & 31;
    if (e >= K_EDGES) return;
    const int s = src[e], d = dst[e], r = rel[e];
    #pragma unroll
    for (int i = 0; i < 12; ++i) {
        const int c = lane + i * 32;
        atomicAdd(&agg[(size_t)d * K_D + c],
                  hn[(size_t)s * K_D + c] + relW[(size_t)r * K_D + c]);
    }
    if (lane == 0) atomicAdd(&cnt[d], 1.0f);
}

// h = LN(relu(hs + agg/max(cnt,1))) ; one wave per node
__global__ void k_sage_finalize(const float* __restrict__ hs, const float* __restrict__ agg,
                                const float* __restrict__ cnt,
                                const float* __restrict__ g, const float* __restrict__ bta,
                                float* __restrict__ h)
{
    const int n    = (blockIdx.x * blockDim.x + threadIdx.x) >> 5;
    const int lane = threadIdx.x & 31;
    if (n >= K_NODES) return;
    const float inv = 1.0f / fmaxf(cnt[n], 1.0f);
    float v[12], s = 0.f;
    #pragma unroll
    for (int i = 0; i < 12; ++i) {
        const int d = lane + i * 32;
        float t = hs[(size_t)n * K_D + d] + agg[(size_t)n * K_D + d] * inv;
        t = fmaxf(t, 0.f);
        v[i] = t; s += t;
    }
    #pragma unroll
    for (int off = 16; off; off >>= 1) s += __shfl_xor(s, off);
    const float mean = s * (1.f / K_D);
    float vs = 0.f;
    #pragma unroll
    for (int i = 0; i < 12; ++i) { const float d = v[i] - mean; vs += d * d; }
    #pragma unroll
    for (int off = 16; off; off >>= 1) vs += __shfl_xor(vs, off);
    const float rstd = rsqrtf(vs * (1.f / K_D) + EPS_LN);
    #pragma unroll
    for (int i = 0; i < 12; ++i) {
        const int d = lane + i * 32;
        h[(size_t)n * K_D + d] = (v[i] - mean) * rstd * g[d] + bta[d];
    }
}

// y = LN(x) ; one wave per row of width 384
__global__ void k_layernorm(const float* __restrict__ x, const float* __restrict__ g,
                            const float* __restrict__ b, float* __restrict__ y, int rows)
{
    const int r    = (blockIdx.x * blockDim.x + threadIdx.x) >> 5;
    const int lane = threadIdx.x & 31;
    if (r >= rows) return;
    float v[12], s = 0.f;
    #pragma unroll
    for (int i = 0; i < 12; ++i) { v[i] = x[(size_t)r * K_D + lane + i * 32]; s += v[i]; }
    #pragma unroll
    for (int off = 16; off; off >>= 1) s += __shfl_xor(s, off);
    const float mean = s * (1.f / K_D);
    float vs = 0.f;
    #pragma unroll
    for (int i = 0; i < 12; ++i) { const float d = v[i] - mean; vs += d * d; }
    #pragma unroll
    for (int off = 16; off; off >>= 1) vs += __shfl_xor(vs, off);
    const float rstd = rsqrtf(vs * (1.f / K_D) + EPS_LN);
    #pragma unroll
    for (int i = 0; i < 12; ++i) {
        const int d = lane + i * 32;
        y[(size_t)r * K_D + d] = (v[i] - mean) * rstd * g[d] + b[d];
    }
}

__global__ void k_build_tokens(const float* __restrict__ h, const float* __restrict__ cls,
                               float* __restrict__ tok)
{
    const size_t i = (size_t)blockIdx.x * 256 + threadIdx.x;
    if (i >= (size_t)K_BT * K_D) return;
    const int d = (int)(i % K_D);
    const size_t bt = i / K_D;
    const int t = (int)(bt % K_TP);
    const size_t b = bt / K_TP;
    float v;
    if (t == 0)            v = cls[d];
    else if (t <= K_TPG)   v = h[(b * K_TPG + (t - 1)) * K_D + d];
    else                   v = 0.f;   // padded rows (masked as keys in attention)
    tok[i] = v;
}

__global__ void k_graph_mean(const float* __restrict__ h, float* __restrict__ out)
{
    const int i = blockIdx.x * 256 + threadIdx.x;      // 64*384 threads
    if (i >= K_B * K_D) return;
    const int d = i % K_D, b = i / K_D;
    float s = 0.f;
    for (int t = 0; t < K_TPG; ++t) s += h[((size_t)b * K_TPG + t) * K_D + d];
    out[i] = s * (1.f / K_TPG);
}

// y = x / max(||x||, 1e-12) ; one wave per row
__global__ void k_row_normalize(const float* __restrict__ x, float* __restrict__ y, int rows)
{
    const int r    = (blockIdx.x * blockDim.x + threadIdx.x) >> 5;
    const int lane = threadIdx.x & 31;
    if (r >= rows) return;
    float v[12], ss = 0.f;
    #pragma unroll
    for (int i = 0; i < 12; ++i) { v[i] = x[(size_t)r * K_D + lane + i * 32]; ss += v[i] * v[i]; }
    #pragma unroll
    for (int off = 16; off; off >>= 1) ss += __shfl_xor(ss, off);
    const float scale = 1.0f / fmaxf(sqrtf(ss), 1e-12f);
    #pragma unroll
    for (int i = 0; i < 12; ++i)
        y[(size_t)r * K_D + lane + i * 32] = v[i] * scale;
}

// =====================================================================
// Host orchestration
// =====================================================================
static void run_gemm(int act, bool res,
                     const float* A, int lda, const float* W, const float* bias,
                     const float* Res, int ldr, float* C, int ldc,
                     int M, int N, int K, hipStream_t st)
{
    dim3 grid(N / GBN, M / GBM), blk(256);
    if (res)          k_gemm<0, true ><<<grid, blk, 0, st>>>(A, lda, W, bias, Res, ldr, C, ldc, M, N, K);
    else if (act == 0) k_gemm<0, false><<<grid, blk, 0, st>>>(A, lda, W, bias, Res, ldr, C, ldc, M, N, K);
    else if (act == 1) k_gemm<1, false><<<grid, blk, 0, st>>>(A, lda, W, bias, Res, ldr, C, ldc, M, N, K);
    else               k_gemm<2, false><<<grid, blk, 0, st>>>(A, lda, W, bias, Res, ldr, C, ldc, M, N, K);
}

struct ConvP {
    const float *Wn, *Wr, *Ws, *bs, *lnb, *lng, *rel;
};

static void run_encoder(const float* x, const int* label, const int* eidx, const int* erel,
                        const float* W_in, const float* b_in, const float* lemb,
                        const ConvP cv[2],
                        float* h, float* x160, float* hn, float* hs,
                        float* agg, float* cnt, float* relW, hipStream_t st)
{
    k_embed_concat<<<(K_NODES * 160) / 256, 256, 0, st>>>(x, label, lemb, x160);
    // h = relu(x160 @ W_in^T + b_in)
    run_gemm(1, false, x160, 160, W_in, b_in, nullptr, 0, h, K_D, K_NODES, K_D, 160, st);
    for (int l = 0; l < 2; ++l) {
        // relW[r] = rel_emb[r] @ W_rel^T
        run_gemm(0, false, cv[l].rel, K_RD, cv[l].Wr, nullptr, nullptr, 0, relW, K_D,
                 K_RB, K_D, K_RD, st);
        // hn = h @ W_neigh^T ; hs = h @ W_self^T + b_self
        run_gemm(0, false, h, K_D, cv[l].Wn, nullptr, nullptr, 0, hn, K_D, K_NODES, K_D, K_D, st);
        run_gemm(0, false, h, K_D, cv[l].Ws, cv[l].bs, nullptr, 0, hs, K_D, K_NODES, K_D, K_D, st);
        hipMemsetAsync(agg, 0, (size_t)K_NODES * K_D * sizeof(float), st);
        hipMemsetAsync(cnt, 0, (size_t)K_NODES * sizeof(float), st);
        k_edge_scatter<<<(K_EDGES * 32) / 256, 256, 0, st>>>(eidx, eidx + K_EDGES, erel,
                                                             hn, relW, agg, cnt);
        k_sage_finalize<<<K_NODES / 8, 256, 0, st>>>(hs, agg, cnt, cv[l].lng, cv[l].lnb, h);
    }
}

extern "C" void kernel_launch(void* const* d_in, const int* in_sizes, int n_in,
                              void* d_out, int out_size, void* d_ws, size_t ws_size,
                              hipStream_t stream)
{
    (void)in_sizes; (void)n_in; (void)out_size; (void)ws_size;
    auto F = [&](int i) { return (const float*)d_in[i]; };
    auto I = [&](int i) { return (const int*)d_in[i]; };

    // ---- inputs (setup_inputs order) ----
    const float* x_ctx   = F(0);  const int* lab_ctx = I(1);
    const int*   ei_ctx  = I(2);  const int* er_ctx  = I(3);
    const float* x_tgt   = F(5);  const int* lab_tgt = I(6);
    const int*   ei_tgt  = I(7);  const int* er_tgt  = I(8);

    // ---- params (jax tree_flatten order: sorted dict keys) ----
    const float* W_in = F(10); const float* b_in = F(11);
    ConvP cv[2];
    for (int l = 0; l < 2; ++l) {
        const int p = 12 + l * 7;
        cv[l].Wn  = F(p + 0); cv[l].Wr  = F(p + 1); cv[l].Ws = F(p + 2);
        cv[l].bs  = F(p + 3); cv[l].lnb = F(p + 4); cv[l].lng = F(p + 5);
        cv[l].rel = F(p + 6);
    }
    const float* lemb  = F(26);
    const float* W_out = F(27); const float* b_out = F(28); const float* cls = F(29);

    // ---- workspace carve ----
    char* ws = (char*)d_ws;
    size_t off = 0;
    auto carve = [&](size_t bytes) { void* p = ws + off; off += (bytes + 255) & ~(size_t)255; return p; };
    float* tokens = (float*)carve((size_t)K_BT * K_D * 4);        // persistent: predictor x
    float* h      = (float*)carve((size_t)K_NODES * K_D * 4);     // encoder output (ctx then tgt)
    float* pool   = (float*)carve((size_t)K_B * K_D * 4);         // pooled rows
    char*  scr    = ws + off;                                     // reused scratch region
    // encoder views
    float* x160 = (float*)(scr);
    float* hn   = (float*)(scr + (((size_t)K_NODES * 160 * 4 + 255) & ~(size_t)255));
    float* hs   = hn   + (size_t)K_NODES * K_D;
    float* agg  = hs   + (size_t)K_NODES * K_D;
    float* cnt  = agg  + (size_t)K_NODES * K_D;
    float* relW = cnt  + K_NODES;
    // predictor views (alias same scratch; encoder is finished before use)
    float* h1   = (float*)scr;
    float* obuf = h1   + (size_t)K_BT * K_D;
    float* big  = obuf + (size_t)K_BT * K_D;      // qkv (1152) and ffn-mid (1536)

    // ================= context encoder =================
    run_encoder(x_ctx, lab_ctx, ei_ctx, er_ctx, W_in, b_in, lemb, cv,
                h, x160, hn, hs, agg, cnt, relW, stream);
    k_build_tokens<<<((size_t)K_BT * K_D) / 256, 256, 0, stream>>>(h, cls, tokens);

    // ================= target encoder + target output =================
    run_encoder(x_tgt, lab_tgt, ei_tgt, er_tgt, W_in, b_in, lemb, cv,
                h, x160, hn, hs, agg, cnt, relW, stream);
    k_graph_mean<<<(K_B * K_D) / 256, 256, 0, stream>>>(h, pool);
    k_row_normalize<<<(K_B + 7) / 8, 256, 0, stream>>>(pool, (float*)d_out + K_B * K_D, K_B);

    // ================= predictor (4 transformer layers) =================
    const int BT = K_BT;
    for (int l = 0; l < 4; ++l) {
        const int p = 30 + 12 * l;
        const float *W1 = F(p+0), *W2 = F(p+1), *Wo = F(p+2), *Wqkv = F(p+3);
        const float *b1 = F(p+4), *b2 = F(p+5), *bo = F(p+6), *bqkv = F(p+7);
        const float *l1b = F(p+8), *l1g = F(p+9), *l2b = F(p+10), *l2g = F(p+11);

        k_layernorm<<<BT / 8, 256, 0, stream>>>(tokens, l1g, l1b, h1, BT);
        run_gemm(0, false, h1, K_D, Wqkv, bqkv, nullptr, 0, big, 1152, BT, 1152, K_D, stream);
        k_attention<<<(K_B * K_NH * 34) / 8, 256, 0, stream>>>(big, obuf);
        run_gemm(0, true, obuf, K_D, Wo, bo, tokens, K_D, tokens, K_D, BT, K_D, K_D, stream);
        k_layernorm<<<BT / 8, 256, 0, stream>>>(tokens, l2g, l2b, h1, BT);
        run_gemm(2, false, h1, K_D, W1, b1, nullptr, 0, big, 1536, BT, 1536, K_D, stream);
        run_gemm(0, true, big, 1536, W2, b2, tokens, K_D, tokens, K_D, BT, K_D, 1536, stream);
    }
    // cls head: rows = token 0 of each graph (strided lda), then normalize
    run_gemm(0, false, tokens, K_TP * K_D, W_out, b_out, nullptr, 0, pool, K_D,
             K_B, K_D, K_D, stream);
    k_row_normalize<<<(K_B + 7) / 8, 256, 0, stream>>>(pool, (float*)d_out, K_B);
}
